// MultiHeadSelfAttention_59742995087536
// MI455X (gfx1250) — matrix operand order repaired and checked by reading, not compiled, not run on hardware
//
#include <hip/hip_runtime.h>
#include <hip/hip_bf16.h>

// ---------------- problem constants ----------------
#define DMODEL 1024
#define NHEADS 16
#define HDIM   64
#define BATCH  4
#define SEQ    2048
#define MROWS  (BATCH * SEQ)   // 8192

// ---------------- vector types ----------------
typedef __attribute__((ext_vector_type(16))) __bf16 v16bf;
typedef __attribute__((ext_vector_type(8)))  __bf16 v8bf;
typedef __attribute__((ext_vector_type(8)))  float  v8f;

union V16U { v16bf v; v8bf h[2]; };

__device__ __forceinline__ unsigned short f2bf(float f) {
    unsigned int u = __float_as_uint(f);
    unsigned int r = u + 0x7fffu + ((u >> 16) & 1u);   // round-to-nearest-even
    return (unsigned short)(r >> 16);
}

__device__ __forceinline__ v16bf ld16(const unsigned short* p) {
    return *reinterpret_cast<const v16bf*>(p);
}
__device__ __forceinline__ v8bf ld8(const unsigned short* p) {
    return *reinterpret_cast<const v8bf*>(p);
}

__device__ __forceinline__ v8f wmma_bf16(v16bf a, v16bf b, v8f c) {
    return __builtin_amdgcn_wmma_f32_16x16x32_bf16(false, a, false, b, (short)0, c,
                                                   false, false);
}

// ---------------- fp32 -> bf16 convert ----------------
__global__ void cvt_f32_bf16(const float* __restrict__ src,
                             unsigned short* __restrict__ dst, int n) {
    int i = blockIdx.x * blockDim.x + threadIdx.x;
    if (i < n) dst[i] = f2bf(src[i]);
}

// ---------------- bf16 WMMA GEMM: C = A @ W^T + bias ----------------
// A: [M,K] bf16 row-major. W: [N,K] bf16 row-major (torch Linear weight).
// Wave tile: 32(M) x 64(N); B operands shared across the two M sub-tiles.
// mode 0: out bf16 [B,H,S,hd]         (Q / K)
// mode 2: out bf16 [B,H,hd,S]         (V transposed)
// mode 3: out fp32 [M,N] row-major    (final projection)
__global__ __launch_bounds__(128, 1)
void gemm_bf16_wmma(const unsigned short* __restrict__ A,
                    const unsigned short* __restrict__ W,
                    const float* __restrict__ bias,
                    void* __restrict__ out,
                    int M, int N, int K, int mode) {
    const int lane = threadIdx.x & 31;
    const int wave = threadIdx.x >> 5;
    const int half = lane >> 4;
    const int l16  = lane & 15;
    const int mrow  = (blockIdx.x * 4 + wave) * 32;
    const int nbase = blockIdx.y * 64;

    v8f acc[2][4] = {};

    const unsigned short* ap0 = A + (size_t)(mrow + l16) * K + half * 8;
    const unsigned short* ap1 = ap0 + (size_t)16 * K;

    for (int k0 = 0; k0 < K; k0 += 32) {
        V16U a0, a1;
        a0.h[0] = ld8(ap0 + k0);
        a0.h[1] = ld8(ap0 + k0 + 16);
        a1.h[0] = ld8(ap1 + k0);
        a1.h[1] = ld8(ap1 + k0 + 16);
#pragma unroll
        for (int nt = 0; nt < 4; ++nt) {
            int ncol = nbase + nt * 16 + l16;
            V16U bu;
            bu.h[0] = ld8(W + (size_t)ncol * K + k0 + half * 8);
            bu.h[1] = ld8(W + (size_t)ncol * K + k0 + half * 8 + 16);
            v16bf b = bu.v;
            acc[0][nt] = wmma_bf16(a0.v, b, acc[0][nt]);
            acc[1][nt] = wmma_bf16(a1.v, b, acc[1][nt]);
        }
    }

#pragma unroll
    for (int mt = 0; mt < 2; ++mt) {
#pragma unroll
        for (int nt = 0; nt < 4; ++nt) {
            int ncol = nbase + nt * 16 + l16;
            float bv = bias[ncol];
            int h = ncol >> 6;       // head
            int d = ncol & 63;       // dim in head
#pragma unroll
            for (int r = 0; r < 8; ++r) {
                int grow = mrow + mt * 16 + r + 8 * half;   // = b*SEQ + s
                float val = acc[mt][nt][r] + bv;
                if (mode == 3) {
                    ((float*)out)[(size_t)grow * N + ncol] = val;
                } else {
                    int b = grow >> 11;             // SEQ = 2048
                    int s = grow & (SEQ - 1);
                    size_t idx;
                    if (mode == 2)  // V transposed: [B,H,hd,S]
                        idx = ((size_t)((b * NHEADS + h) * HDIM + d)) * SEQ + s;
                    else            // Q/K: [B,H,S,hd]
                        idx = ((size_t)((b * NHEADS + h) * SEQ + s)) * HDIM + d;
                    ((unsigned short*)out)[idx] = f2bf(val);
                }
            }
        }
    }
}

// ---------------- flash attention (causal, online softmax) ----------------
// Q,K: [B,H,S,hd] bf16.  Vt: [B,H,hd,S] bf16.  Ctx out: [B,S,H,hd] bf16.
// Block = 4 waves = 64 queries; K/V chunks staged once per block in LDS.
__global__ __launch_bounds__(128, 1)
void flash_attn_wmma(const unsigned short* __restrict__ Q,
                     const unsigned short* __restrict__ Kmat,
                     const unsigned short* __restrict__ Vt,
                     unsigned short* __restrict__ Ctx) {
    // K chunk: 32 keys x 64 d, row stride 72 (144B: 16B aligned, 36-bank step,
    // conflict-free over 16 consecutive rows). V chunk: 64 d x 32 keys,
    // row stride 40 (80B). pbuf: per-wave P re-layout, row stride 40.
    __shared__ __align__(16) unsigned short ldsK[32][72];
    __shared__ __align__(16) unsigned short ldsV[64][40];
    __shared__ __align__(16) unsigned short pbuf[4][16][40];

    const int tid  = threadIdx.x;
    const int lane = tid & 31;
    const int wave = tid >> 5;
    const int half = lane >> 4;
    const int l16  = lane & 15;

    const int bh        = blockIdx.y;                  // b*NHEADS + h
    const int qbase_blk = blockIdx.x * 64;
    const int qbase     = qbase_blk + wave * 16;

    const unsigned short* Qp = Q    + (size_t)bh * SEQ * HDIM;
    const unsigned short* Kp = Kmat + (size_t)bh * SEQ * HDIM;
    const unsigned short* Vp = Vt   + (size_t)bh * HDIM * SEQ;

    // Q tile 16x64 in A-layout, split into two K=32 operands
    V16U qa[2];
    {
        const unsigned short* qr = Qp + (size_t)(qbase + l16) * HDIM + half * 8;
        qa[0].h[0] = ld8(qr);      qa[0].h[1] = ld8(qr + 16);
        qa[1].h[0] = ld8(qr + 32); qa[1].h[1] = ld8(qr + 48);
    }

    v8f acc[4] = {};
    float rmax[8], rsum[8];
#pragma unroll
    for (int r = 0; r < 8; ++r) { rmax[r] = -__builtin_inff(); rsum[r] = 0.f; }

    const float sscale = 0.125f;   // 1/sqrt(64)
    const int blocklast = qbase_blk + 63;   // block-level causal bound

    for (int kk = 0; kk <= blocklast; kk += 32) {
        // ---- cooperative stage of K chunk (32x64) and V chunk (64x32) ----
        {
            int row = tid >> 2, cc = (tid & 3) << 4;     // K: 32 rows x 4 chunks
            int kg = kk + row; if (kg > SEQ - 1) kg = SEQ - 1;
            const unsigned short* src = Kp + (size_t)kg * HDIM + cc;
            *(v8bf*)&ldsK[row][cc]     = ld8(src);
            *(v8bf*)&ldsK[row][cc + 8] = ld8(src + 8);

            int d = tid >> 1, c2 = (tid & 1) << 4;       // V: 64 rows x 2 chunks
            int col = kk + c2; if (col > SEQ - 16) col = SEQ - 16;
            const unsigned short* vs = Vp + (size_t)d * SEQ + col;
            *(v8bf*)&ldsV[d][c2]     = ld8(vs);
            *(v8bf*)&ldsV[d][c2 + 8] = ld8(vs + 8);
        }
        __syncthreads();

        // ---- scores: S[16q x 32k] = Q @ K^T (two 16x16 tiles, K-dim 64) ----
        v8f st[2];
#pragma unroll
        for (int t = 0; t < 2; ++t) {
            int krow = t * 16 + l16;
            V16U b0, b1;
            b0.h[0] = *(const v8bf*)&ldsK[krow][half * 8];
            b0.h[1] = *(const v8bf*)&ldsK[krow][half * 8 + 16];
            b1.h[0] = *(const v8bf*)&ldsK[krow][32 + half * 8];
            b1.h[1] = *(const v8bf*)&ldsK[krow][32 + half * 8 + 16];
            v8f z = {};
            z = wmma_bf16(qa[0].v, b0.v, z);
            st[t] = wmma_bf16(qa[1].v, b1.v, z);
        }

        // ---- scale + causal mask + chunk row-max ----
        float cm[8];
#pragma unroll
        for (int r = 0; r < 8; ++r) {
            int mq = qbase + r + 8 * half;
            float v0 = st[0][r] * sscale;
            float v1 = st[1][r] * sscale;
            if (kk + l16 > mq)      v0 = -__builtin_inff();
            if (kk + 16 + l16 > mq) v1 = -__builtin_inff();
            st[0][r] = v0; st[1][r] = v1;
            cm[r] = fmaxf(v0, v1);
        }
#pragma unroll
        for (int r = 0; r < 8; ++r)
#pragma unroll
            for (int m = 1; m < 16; m <<= 1)
                cm[r] = fmaxf(cm[r], __shfl_xor(cm[r], m, 32));

        // ---- online softmax update (fully-masked chunk: corr=1, p=0) ----
        float csum[8];
#pragma unroll
        for (int r = 0; r < 8; ++r) {
            float nm   = fmaxf(rmax[r], cm[r]);
            float corr = __expf(rmax[r] - nm);
            rmax[r] = nm;
            float p0 = __expf(st[0][r] - nm);
            float p1 = __expf(st[1][r] - nm);
            int m = r + 8 * half;
            pbuf[wave][m][l16]      = f2bf(p0);
            pbuf[wave][m][16 + l16] = f2bf(p1);
            csum[r] = p0 + p1;
#pragma unroll
            for (int dt = 0; dt < 4; ++dt) acc[dt][r] *= corr;
            rsum[r] *= corr;
        }
#pragma unroll
        for (int r = 0; r < 8; ++r) {
#pragma unroll
            for (int m = 1; m < 16; m <<= 1)
                csum[r] += __shfl_xor(csum[r], m, 32);
            rsum[r] += csum[r];
        }

        // ---- re-read P in A-layout from LDS (same wave, DS ops in-order) ----
        V16U pa;
        pa.h[0] = *(const v8bf*)&pbuf[wave][l16][half * 8];
        pa.h[1] = *(const v8bf*)&pbuf[wave][l16][half * 8 + 16];

        // ---- ctx += P @ V (4 d-tiles of 16), B operand from LDS ----
#pragma unroll
        for (int dt = 0; dt < 4; ++dt) {
            int drow = dt * 16 + l16;
            V16U vb;
            vb.h[0] = *(const v8bf*)&ldsV[drow][half * 8];
            vb.h[1] = *(const v8bf*)&ldsV[drow][half * 8 + 16];
            acc[dt] = wmma_bf16(pa.v, vb.v, acc[dt]);
        }
        __syncthreads();   // protect LDS K/V before next-chunk restage
    }

    // ---- epilogue: ctx[b, s, h, d] = acc / rowsum ----
    const int b = bh >> 4;
    const int h = bh & 15;
#pragma unroll
    for (int dt = 0; dt < 4; ++dt) {
        int d = h * HDIM + dt * 16 + l16;
#pragma unroll
        for (int r = 0; r < 8; ++r) {
            int s = qbase + r + 8 * half;
            float val = acc[dt][r] / rsum[r];
            Ctx[((size_t)(b * SEQ + s)) * DMODEL + d] = f2bf(val);
        }
    }
}

// ---------------- host orchestration ----------------
extern "C" void kernel_launch(void* const* d_in, const int* in_sizes, int n_in,
                              void* d_out, int out_size, void* d_ws, size_t ws_size,
                              hipStream_t stream) {
    // inputs: 0 emb, 1 mask(ignored; causal computed analytically),
    //         2 Wq, 3 bq, 4 Wk, 5 bk, 6 Wv, 7 bv, 8 Wo, 9 bo
    const float* emb = (const float*)d_in[0];
    const float* Wq  = (const float*)d_in[2];
    const float* bq  = (const float*)d_in[3];
    const float* Wk  = (const float*)d_in[4];
    const float* bk  = (const float*)d_in[5];
    const float* Wv  = (const float*)d_in[6];
    const float* bv  = (const float*)d_in[7];
    const float* Wo  = (const float*)d_in[8];
    const float* bo  = (const float*)d_in[9];

    // workspace layout (88 MB total, bf16):
    //   Xb 16MB | Wq/Wk/Wv/Wo 4x2MB | Q 16MB | K 16MB | Vt 16MB | Ctx 16MB
    char* ws = (char*)d_ws;
    unsigned short* Xb   = (unsigned short*)ws;
    unsigned short* Wqb  = (unsigned short*)(ws + ((size_t)16 << 20));
    unsigned short* Wkb  = Wqb + (size_t)DMODEL * DMODEL;
    unsigned short* Wvb  = Wkb + (size_t)DMODEL * DMODEL;
    unsigned short* Wob  = Wvb + (size_t)DMODEL * DMODEL;
    unsigned short* Qb   = (unsigned short*)(ws + ((size_t)24 << 20));
    unsigned short* Kb   = Qb + (size_t)MROWS * DMODEL;
    unsigned short* Vtb  = Kb + (size_t)MROWS * DMODEL;
    unsigned short* Ctxb = Vtb + (size_t)MROWS * DMODEL;

    const int nX = MROWS * DMODEL;          // 8388608
    const int nW = DMODEL * DMODEL;         // 1048576

    cvt_f32_bf16<<<(nX + 255) / 256, 256, 0, stream>>>(emb, Xb, nX);
    cvt_f32_bf16<<<(nW + 255) / 256, 256, 0, stream>>>(Wq, Wqb, nW);
    cvt_f32_bf16<<<(nW + 255) / 256, 256, 0, stream>>>(Wk, Wkb, nW);
    cvt_f32_bf16<<<(nW + 255) / 256, 256, 0, stream>>>(Wv, Wvb, nW);
    cvt_f32_bf16<<<(nW + 255) / 256, 256, 0, stream>>>(Wo, Wob, nW);

    dim3 gg(MROWS / 128, DMODEL / 64);      // (64, 16)
    gemm_bf16_wmma<<<gg, 128, 0, stream>>>(Xb, Wqb, bq, Qb,  MROWS, DMODEL, DMODEL, 0);
    gemm_bf16_wmma<<<gg, 128, 0, stream>>>(Xb, Wkb, bk, Kb,  MROWS, DMODEL, DMODEL, 0);
    gemm_bf16_wmma<<<gg, 128, 0, stream>>>(Xb, Wvb, bv, Vtb, MROWS, DMODEL, DMODEL, 2);

    flash_attn_wmma<<<dim3(SEQ / 64, BATCH * NHEADS), 128, 0, stream>>>(Qb, Kb, Vtb, Ctxb);

    gemm_bf16_wmma<<<gg, 128, 0, stream>>>(Ctxb, Wob, bo, d_out, MROWS, DMODEL, DMODEL, 3);
}
